// RBF_91122026152887
// MI455X (gfx1250) — compile-verified, matrix operand order checked by reading, not run on hardware
//
#include <hip/hip_runtime.h>
#include <hip/hip_bf16.h>

// RBF pairwise squared distances on MI455X (gfx1250):
//   out[n,m] = ||x_n||^2 + ||c_m||^2 - 2 * <x_n, c_m>
// Cross term via WMMA bf16 with 2-term bf16 split (hi+lo) for ~f32 accuracy:
//   x*c ~= hi(x)hi(c) + hi(x)lo(c) + lo(x)hi(c), accumulated in f32.
// Global->LDS path is software-pipelined in registers: chunk k+1's f32 tiles
// are prefetched from global while chunk k's 48 WMMAs execute.

typedef __attribute__((ext_vector_type(16))) __bf16 v16bf;
typedef __attribute__((ext_vector_type(8)))  float  v8f;

#define N_ROWS 16384
#define M_COLS 4096
#define DDIM   512
#define KC     32          // K elements per LDS chunk
#define NCHUNK (DDIM / KC) // 16
#define BROWS  128         // workgroup output rows (input rows)
#define BCOLS  256         // workgroup output cols (centre rows)
#define KSTRIDE_B 80       // LDS bytes per row: 40 halves (32 used + pad),
                           // 16B-aligned, bank-conflict-free b128 frag loads

union Frag {
    unsigned int u[8];
    uint4        q[2];
    v16bf        v;
};

static __device__ inline void split2(float x, unsigned short& hb, unsigned short& lb) {
    __bf16 h = (__bf16)x;            // RNE f32 -> bf16
    float  hf = (float)h;
    __bf16 l = (__bf16)(x - hf);     // residual, next 8 mantissa bits
    union { __bf16 b; unsigned short s; } u;
    u.b = h; hb = u.s;
    u.b = l; lb = u.s;
}

static __device__ inline void cvt_store4(char* hiP, char* loP, float4 f) {
    unsigned short h0, l0, h1, l1, h2, l2, h3, l3;
    split2(f.x, h0, l0);
    split2(f.y, h1, l1);
    split2(f.z, h2, l2);
    split2(f.w, h3, l3);
    uint2 hw, lw;
    hw.x = (unsigned)h0 | ((unsigned)h1 << 16);
    hw.y = (unsigned)h2 | ((unsigned)h3 << 16);
    lw.x = (unsigned)l0 | ((unsigned)l1 << 16);
    lw.y = (unsigned)l2 | ((unsigned)l3 << 16);
    *(uint2*)hiP = hw;               // ds_store_b64 (8B aligned)
    *(uint2*)loP = lw;
}

// One wave (32 lanes) reduces one row of 512 floats -> sum of squares.
__global__ __launch_bounds__(256)
void rbf_norms_kernel(const float* __restrict__ input,
                      const float* __restrict__ centres,
                      float* __restrict__ xsq,
                      float* __restrict__ csq) {
    int wave = (blockIdx.x * blockDim.x + threadIdx.x) >> 5;
    int lane = threadIdx.x & 31;
    const int rowsTotal = N_ROWS + M_COLS;
    if (wave >= rowsTotal) return;
    const float* src = (wave < N_ROWS)
                         ? input   + (size_t)wave * DDIM
                         : centres + (size_t)(wave - N_ROWS) * DDIM;
    float s = 0.0f;
#pragma unroll
    for (int i = 0; i < DDIM / 32; ++i) {
        float v = src[i * 32 + lane];
        s += v * v;
    }
#pragma unroll
    for (int off = 16; off > 0; off >>= 1)
        s += __shfl_down(s, off, 32);
    if (lane == 0) {
        if (wave < N_ROWS) xsq[wave] = s;
        else               csq[wave - N_ROWS] = s;
    }
}

__global__ __launch_bounds__(256)
void rbf_gemm_kernel(const float* __restrict__ input,
                     const float* __restrict__ centres,
                     const float* __restrict__ xsq,
                     const float* __restrict__ csq,
                     float* __restrict__ out) {
    __shared__ __align__(16) char smem[(2 * BROWS + 2 * BCOLS) * KSTRIDE_B]; // 60 KB
    char* aHi = smem;
    char* aLo = aHi + BROWS * KSTRIDE_B;
    char* bHi = aLo + BROWS * KSTRIDE_B;
    char* bLo = bHi + BCOLS * KSTRIDE_B;

    const int tid  = threadIdx.x;
    const int lane = tid & 31;
    const int wid  = tid >> 5;       // 0..7
    const int wr   = wid >> 2;       // 0..1: 64-row group within 128
    const int wc   = wid & 3;        // 0..3: 64-col group within 256
    const int lrow = lane & 15;      // row/col within 16x16 tile
    const int half = lane >> 4;      // A/B K-half selector per ISA layout

    const int rowBlock = blockIdx.y * BROWS;
    const int colBlock = blockIdx.x * BCOLS;

    v8f acc[4][4];
#pragma unroll
    for (int i = 0; i < 4; ++i)
#pragma unroll
        for (int j = 0; j < 4; ++j)
            acc[i][j] = 0.0f;

    // Per-thread staging addresses: idx = tid + 256*i  =>  row = tid/8 + 32*i,
    // c4 = tid&7 (loop-invariant).
    const int r0 = tid >> 3;
    const int c4 = tid & 7;
    const float* aP = input   + (size_t)rowBlock * DDIM + (size_t)r0 * DDIM + c4 * 4;
    const float* bP = centres + (size_t)colBlock * DDIM + (size_t)r0 * DDIM + c4 * 4;
    char* aHiP = aHi + r0 * KSTRIDE_B + c4 * 8;
    char* aLoP = aLo + r0 * KSTRIDE_B + c4 * 8;
    char* bHiP = bHi + r0 * KSTRIDE_B + c4 * 8;
    char* bLoP = bLo + r0 * KSTRIDE_B + c4 * 8;

    // Prologue: prefetch chunk 0 into registers.
    float4 aReg[4], bReg[8];
#pragma unroll
    for (int i = 0; i < 4; ++i)
        aReg[i] = *(const float4*)(aP + (size_t)(32 * i) * DDIM);
#pragma unroll
    for (int i = 0; i < 8; ++i)
        bReg[i] = *(const float4*)(bP + (size_t)(32 * i) * DDIM);

    for (int kc = 0; kc < NCHUNK; ++kc) {
        __syncthreads();  // WAR: previous iter's frag reads vs new stores
        // Drain staged registers -> LDS with f32 -> (bf16 hi, bf16 lo) split.
#pragma unroll
        for (int i = 0; i < 4; ++i)
            cvt_store4(aHiP + 32 * i * KSTRIDE_B, aLoP + 32 * i * KSTRIDE_B, aReg[i]);
#pragma unroll
        for (int i = 0; i < 8; ++i)
            cvt_store4(bHiP + 32 * i * KSTRIDE_B, bLoP + 32 * i * KSTRIDE_B, bReg[i]);
        __syncthreads();

        // Prefetch next chunk from global; latency hides behind the WMMAs.
        if (kc + 1 < NCHUNK) {
            const int ko = (kc + 1) * KC;
#pragma unroll
            for (int i = 0; i < 4; ++i)
                aReg[i] = *(const float4*)(aP + (size_t)(32 * i) * DDIM + ko);
#pragma unroll
            for (int i = 0; i < 8; ++i)
                bReg[i] = *(const float4*)(bP + (size_t)(32 * i) * DDIM + ko);
        }

        // Fragments per ISA 16-bit A 16x32 layout:
        //   lanes 0-15 : K0-7  (bytes 0..15)  and K16-23 (bytes 32..47)
        //   lanes 16-31: K8-15 (bytes 16..31) and K24-31 (bytes 48..63)
        Frag fAhi[4], fAlo[4], fBhi[4], fBlo[4];
#pragma unroll
        for (int i = 0; i < 4; ++i) {
            int ar = wr * 64 + i * 16 + lrow;
            const char* pa = aHi + ar * KSTRIDE_B + half * 16;
            fAhi[i].q[0] = *(const uint4*)(pa);
            fAhi[i].q[1] = *(const uint4*)(pa + 32);
            const char* pl = aLo + ar * KSTRIDE_B + half * 16;
            fAlo[i].q[0] = *(const uint4*)(pl);
            fAlo[i].q[1] = *(const uint4*)(pl + 32);
            int br = wc * 64 + i * 16 + lrow;
            const char* pb = bHi + br * KSTRIDE_B + half * 16;
            fBhi[i].q[0] = *(const uint4*)(pb);
            fBhi[i].q[1] = *(const uint4*)(pb + 32);
            const char* pq = bLo + br * KSTRIDE_B + half * 16;
            fBlo[i].q[0] = *(const uint4*)(pq);
            fBlo[i].q[1] = *(const uint4*)(pq + 32);
        }

        // 48 WMMAs per chunk: hi*hi + hi*lo + lo*hi (drop lo*lo, ~2^-16 rel)
#pragma unroll
        for (int i = 0; i < 4; ++i)
#pragma unroll
            for (int j = 0; j < 4; ++j) {
                acc[i][j] = __builtin_amdgcn_wmma_f32_16x16x32_bf16(
                    false, fAhi[i].v, false, fBhi[j].v, (short)0, acc[i][j], false, false);
                acc[i][j] = __builtin_amdgcn_wmma_f32_16x16x32_bf16(
                    false, fAhi[i].v, false, fBlo[j].v, (short)0, acc[i][j], false, false);
                acc[i][j] = __builtin_amdgcn_wmma_f32_16x16x32_bf16(
                    false, fAlo[i].v, false, fBhi[j].v, (short)0, acc[i][j], false, false);
            }
    }

    // Epilogue: out = xsq + csq - 2*cross.
    // C/D layout: n = lane%16, m = vgpr + 8*(lane/16).
    float cs[4];
#pragma unroll
    for (int j = 0; j < 4; ++j)
        cs[j] = csq[colBlock + wc * 64 + j * 16 + lrow];

#pragma unroll
    for (int i = 0; i < 4; ++i) {
#pragma unroll
        for (int v = 0; v < 8; ++v) {
            int gr = rowBlock + wr * 64 + i * 16 + v + 8 * half;
            float xs = xsq[gr];
#pragma unroll
            for (int j = 0; j < 4; ++j) {
                int gc = colBlock + wc * 64 + j * 16 + lrow;
                out[(size_t)gr * M_COLS + gc] = xs + cs[j] - 2.0f * acc[i][j][v];
            }
        }
    }
}

extern "C" void kernel_launch(void* const* d_in, const int* in_sizes, int n_in,
                              void* d_out, int out_size, void* d_ws, size_t ws_size,
                              hipStream_t stream) {
    const float* input   = (const float*)d_in[0];
    const float* centres = (const float*)d_in[1];
    float* out = (float*)d_out;

    float* xsq = (float*)d_ws;        // 16384 floats
    float* csq = xsq + N_ROWS;        // 4096 floats  (80 KB total in d_ws)

    {
        const int rowsTotal = N_ROWS + M_COLS;      // one wave per row
        const int wavesPerBlock = 256 / 32;
        const int blocks = (rowsTotal + wavesPerBlock - 1) / wavesPerBlock;
        rbf_norms_kernel<<<blocks, 256, 0, stream>>>(input, centres, xsq, csq);
    }

    dim3 grid(M_COLS / BCOLS, N_ROWS / BROWS);      // 16 x 128 workgroups
    rbf_gemm_kernel<<<grid, 256, 0, stream>>>(input, centres, xsq, csq, out);
}
